// WindowGNN_16415365005924
// MI455X (gfx1250) — matrix operand order repaired and checked by reading, not compiled, not run on hardware
//
#include <hip/hip_runtime.h>
#include <hip/hip_bf16.h>
#include <hip/hip_fp16.h>

typedef __attribute__((ext_vector_type(16))) _Float16 v16h;
typedef __attribute__((ext_vector_type(8)))  float    v8f;
typedef int v4i __attribute__((vector_size(16)));

#define TB 256

// ---------------------------------------------------------------- utilities
__global__ void zero_f32_kernel(float* __restrict__ p, long n) {
  long t = (long)blockIdx.x * blockDim.x + threadIdx.x;
  if (t < n) p[t] = 0.0f;
}

// h0[i,j] = relu(x[i]*fcW[j] + fcb[j])   (IN_DIM == 1 -> outer product)
__global__ void embed_kernel(const float* __restrict__ x, const float* __restrict__ fcW,
                             const float* __restrict__ fcb, float* __restrict__ h, long total) {
  long t = (long)blockIdx.x * blockDim.x + threadIdx.x;
  if (t >= total) return;
  long i = t >> 6; int j = (int)(t & 63);
  h[t] = fmaxf(fmaf(x[i], fcW[j], fcb[j]), 0.0f);
}

// deg[dst[e]] += 1
__global__ void deg_kernel(const int* __restrict__ dst, float* __restrict__ deg, long E) {
  long t = (long)blockIdx.x * blockDim.x + threadIdx.x;
  if (t < E) atomicAdd(&deg[dst[t]], 1.0f);
}

// dinv[i] = rsqrt(deg[i] + 1)   (in place)
__global__ void dinv_kernel(float* __restrict__ d, int n) {
  int t = blockIdx.x * blockDim.x + threadIdx.x;
  if (t < n) d[t] = rsqrtf(d[t] + 1.0f);
}

// ------------------------------------------------- WMMA GEMM:  D = A(Nx64) * W(64x64)
// One wave computes a 16x64 output tile via 8x v_wmma_f32_16x16x32_f16 (K=64 -> 2 steps).
// W is staged into LDS with the gfx1250 async global->LDS copier, then transposed to
// f16 Wt[n][k] so each lane's B fragment is one contiguous 32-byte LDS read.
__global__ void gemm64_wmma_kernel(const float* __restrict__ A, const float* __restrict__ W,
                                   float* __restrict__ D, int nRows) {
  __shared__ float    Wstage[64 * 64];   // 16 KB raw f32 copy (async-filled)
  __shared__ _Float16 Wt[64 * 64];       // 8 KB transposed f16: Wt[n*64 + k]
  const int tid = threadIdx.x;

#if __has_builtin(__builtin_amdgcn_global_load_async_to_lds_b128)
  // 256 threads x 4 issues x 16B = 16 KB, VGPR-bypassing async copy (ASYNCcnt)
#pragma unroll
  for (int it = 0; it < 4; ++it) {
    const int idx = (it * TB + tid) * 4;  // float index, 16B granules
    __builtin_amdgcn_global_load_async_to_lds_b128(
        (__attribute__((address_space(1))) v4i*)(W + idx),
        (__attribute__((address_space(3))) v4i*)(Wstage + idx), 0, 0);
  }
#if __has_builtin(__builtin_amdgcn_s_wait_asynccnt)
  __builtin_amdgcn_s_wait_asynccnt(0);
#else
  asm volatile("s_wait_asynccnt 0x0" ::: "memory");
#endif
#else
  for (int i = tid * 4; i < 64 * 64; i += TB * 4)
    *(float4*)(Wstage + i) = *(const float4*)(W + i);
#endif
  __syncthreads();

  // transpose + convert once per block
  for (int i = tid; i < 64 * 64; i += TB) {
    const int k = i >> 6, n = i & 63;
    Wt[n * 64 + k] = (_Float16)Wstage[i];
  }
  __syncthreads();

  const int lane = tid & 31;
  const int wid  = tid >> 5;                       // 8 waves / block
  const int row0 = (blockIdx.x * 8 + wid) * 16;
  if (row0 + 16 > nRows) return;                   // wave-uniform guard (EXEC all-ones inside)

  const int m  = lane & 15;                        // row / col within tile
  const int kh = lane >> 4;                        // lane half selects K sub-range

  // B fragments: 32x16 f16, lane = column n, elem e -> K = kt*32 + kh*16 + e
  // contiguous in Wt -> 2x ds_load_b128 per fragment
  v16h bf[2][4];
#pragma unroll
  for (int kt = 0; kt < 2; ++kt)
#pragma unroll
    for (int nt = 0; nt < 4; ++nt)
      bf[kt][nt] = *(const v16h*)&Wt[(nt * 16 + m) * 64 + kt * 32 + kh * 16];

  v8f acc[4] = { {0,0,0,0,0,0,0,0}, {0,0,0,0,0,0,0,0},
                 {0,0,0,0,0,0,0,0}, {0,0,0,0,0,0,0,0} };

  const float* Ar = A + (size_t)(row0 + m) * 64;
#pragma unroll
  for (int kt = 0; kt < 2; ++kt) {
    // A k order used here (contiguous, same as B): lanes 0-15 hold K = base+0..15; lanes 16-31 hold K = base+16..31
    const int b0 = kt * 32 + kh * 16;
    const int b1 = b0 + 8;
    const float4 u0 = *(const float4*)(Ar + b0);
    const float4 u1 = *(const float4*)(Ar + b0 + 4);
    const float4 u2 = *(const float4*)(Ar + b1);
    const float4 u3 = *(const float4*)(Ar + b1 + 4);
    v16h a;
    a[0]=(_Float16)u0.x; a[1]=(_Float16)u0.y; a[2]=(_Float16)u0.z; a[3]=(_Float16)u0.w;
    a[4]=(_Float16)u1.x; a[5]=(_Float16)u1.y; a[6]=(_Float16)u1.z; a[7]=(_Float16)u1.w;
    a[8]=(_Float16)u2.x; a[9]=(_Float16)u2.y; a[10]=(_Float16)u2.z; a[11]=(_Float16)u2.w;
    a[12]=(_Float16)u3.x; a[13]=(_Float16)u3.y; a[14]=(_Float16)u3.z; a[15]=(_Float16)u3.w;
#pragma unroll
    for (int nt = 0; nt < 4; ++nt)
      acc[nt] = __builtin_amdgcn_wmma_f32_16x16x32_f16(
          /*neg_a=*/false, a, /*neg_b=*/false, bf[kt][nt],
          /*c_mod=*/(short)0, acc[nt], /*reuse_a=*/false, /*reuse_b=*/false);
  }

  // D layout: VGPR r -> row = row0 + kh*8 + r, col = nt*16 + m
#pragma unroll
  for (int r = 0; r < 8; ++r) {
    const size_t row = (size_t)(row0 + kh * 8 + r);
#pragma unroll
    for (int nt = 0; nt < 4; ++nt)
      D[row * 64 + nt * 16 + m] = acc[nt][r];
  }
}

// --------------------------------------------- edge aggregation (L2-resident scatter)
// one wave per edge; lane owns 2 consecutive floats of the 64-wide feature row
__global__ void edge_scatter_kernel(const int* __restrict__ src, const int* __restrict__ dst,
                                    const float* __restrict__ dinv, const float* __restrict__ hw,
                                    float* __restrict__ agg, long E) {
  const long wave = ((long)blockIdx.x * blockDim.x + threadIdx.x) >> 5;
  const int lane = threadIdx.x & 31;
  if (wave >= E) return;
  const int s = src[wave], d = dst[wave];
  const float w = dinv[s] * dinv[d];
  const int j = lane * 2;
  const float2 v = *(const float2*)(hw + (size_t)s * 64 + j);
  atomicAdd(agg + (size_t)d * 64 + j,     v.x * w);
  atomicAdd(agg + (size_t)d * 64 + j + 1, v.y * w);
}

// h_next = relu(agg + hw*dinv[i]^2 + b[j])     (agg may alias h_next elementwise)
__global__ void finalize_kernel(const float* __restrict__ agg, const float* __restrict__ hw,
                                const float* __restrict__ dinv, const float* __restrict__ bias,
                                float* __restrict__ hout, long total) {
  long t = (long)blockIdx.x * blockDim.x + threadIdx.x;
  if (t >= total) return;
  long i = t >> 6; int j = (int)(t & 63);
  const float di = dinv[i];
  hout[t] = fmaxf(agg[t] + hw[t] * di * di + bias[j], 0.0f);
}

// -------------------------------------------------------------- global mean pool
__global__ void pool_sum_kernel(const float* __restrict__ h, const int* __restrict__ batch,
                                float* __restrict__ out, float* __restrict__ cnt, long total) {
  long t = (long)blockIdx.x * blockDim.x + threadIdx.x;
  if (t >= total) return;
  long i = t >> 6; int j = (int)(t & 63);
  const int g = batch[i];
  atomicAdd(out + (size_t)g * 64 + j, h[t]);
  if (j == 0) atomicAdd(cnt + g, 1.0f);
}

__global__ void pool_div_kernel(float* __restrict__ out, const float* __restrict__ cnt, int total) {
  int t = blockIdx.x * blockDim.x + threadIdx.x;
  if (t >= total) return;
  out[t] /= fmaxf(cnt[t >> 6], 1.0f);
}

// ------------------------------------------------------------------ driver
extern "C" void kernel_launch(void* const* d_in, const int* in_sizes, int n_in,
                              void* d_out, int out_size, void* d_ws, size_t ws_size,
                              hipStream_t stream) {
  const float* x    = (const float*)d_in[0];
  const int*   ei   = (const int*)  d_in[1];
  const int*   bat  = (const int*)  d_in[2];
  const float* fcW  = (const float*)d_in[3];
  const float* fcb  = (const float*)d_in[4];
  const float* W1   = (const float*)d_in[5];
  const float* b1   = (const float*)d_in[6];
  const float* W2   = (const float*)d_in[7];
  const float* b2   = (const float*)d_in[8];
  float* out = (float*)d_out;

  const int  N = in_sizes[0];               // 250000 nodes
  const long E = (long)in_sizes[1] / 2;     // 4000000 edges
  const int  G64 = out_size;                // 512*64

  char* w = (char*)d_ws;
  const size_t NB = (size_t)N * 64 * sizeof(float);       // 64 MB per feature buffer
  float* P0   = (float*)(w);
  float* P1   = (float*)(w + NB);
  float* P2   = (float*)(w + 2 * NB);
  float* dinv = (float*)(w + 3 * NB);
  float* cnt  = (float*)(w + 3 * NB + (((size_t)N * 4 + 255) & ~(size_t)255));

  const int* srcI = ei;
  const int* dstI = ei + E;

  const long NE64 = (long)N * 64;
  const int gElem  = (int)((NE64 + TB - 1) / TB);
  const int gNode  = (N + TB - 1) / TB;
  const int gEdgeT = (int)((E + TB - 1) / TB);        // thread-per-edge
  const int gEdgeW = (int)((E + 7) / 8);              // wave-per-edge (8 waves/block)
  const int gGemm  = (N / 16 + 7) / 8;                // 16-row tile per wave, 8 waves/block

  // degrees -> dinv (shared by both convs)
  zero_f32_kernel<<<gNode, TB, 0, stream>>>(dinv, N);
  deg_kernel<<<gEdgeT, TB, 0, stream>>>(dstI, dinv, E);
  dinv_kernel<<<gNode, TB, 0, stream>>>(dinv, N);

  // embed
  embed_kernel<<<gElem, TB, 0, stream>>>(x, fcW, fcb, P0, NE64);

  // conv1: P0 -> P2
  gemm64_wmma_kernel<<<gGemm, TB, 0, stream>>>(P0, W1, P1, N);
  zero_f32_kernel<<<gElem, TB, 0, stream>>>(P2, NE64);
  edge_scatter_kernel<<<gEdgeW, TB, 0, stream>>>(srcI, dstI, dinv, P1, P2, E);
  finalize_kernel<<<gElem, TB, 0, stream>>>(P2, P1, dinv, b1, P2, NE64);

  // conv2: P2 -> P0
  gemm64_wmma_kernel<<<gGemm, TB, 0, stream>>>(P2, W2, P1, N);
  zero_f32_kernel<<<gElem, TB, 0, stream>>>(P0, NE64);
  edge_scatter_kernel<<<gEdgeW, TB, 0, stream>>>(srcI, dstI, dinv, P1, P0, E);
  finalize_kernel<<<gElem, TB, 0, stream>>>(P0, P1, dinv, b2, P0, NE64);

  // global mean pool -> out
  zero_f32_kernel<<<(G64 + TB - 1) / TB, TB, 0, stream>>>(out, G64);
  zero_f32_kernel<<<1, TB, 0, stream>>>(cnt, G64 / 64);
  pool_sum_kernel<<<gElem, TB, 0, stream>>>(P0, bat, out, cnt, NE64);
  pool_div_kernel<<<(G64 + TB - 1) / TB, TB, 0, stream>>>(out, cnt, G64);
}
